// VQTokenizerHead_70918499991688
// MI455X (gfx1250) — matrix operand order repaired and checked by reading, not compiled, not run on hardware
//
#include <hip/hip_runtime.h>

// MI455X / gfx1250, wave32, WMMA f16->f32 path.
typedef __attribute__((ext_vector_type(16))) _Float16 v16h;
typedef __attribute__((ext_vector_type(8)))  float    v8f;

#define N_ROWS      (64 * 2048)    // 131072 flattened tokens
#define DDIM        256
#define KCODES      512
#define ROWS_PER_WG 128
#define WAVES       8
#define THREADS     256
#define CHUNK_COLS  128
#define NCHUNK      (KCODES / CHUNK_COLS)   // 4
#define PITCH       264            // f16 per LDS row (256 + 8 pad to break bank conflicts)

// Flat output layout (reference tuple order): z_q_st, indices, commit_loss, z_q_detached
#define ZQST_OFF 0
#define IDX_OFF  (N_ROWS * DDIM)
#define LOSS_OFF (IDX_OFF + N_ROWS)
#define ZQD_OFF  (LOSS_OFF + 1)

// LDS carve offsets (bytes)
#define LDS_ENORM 0
#define LDS_A     2048
#define LDS_B     (LDS_A + ROWS_PER_WG * PITCH * 2)
#define LDS_IDX   (LDS_B + ROWS_PER_WG * PITCH * 2)
#define LDS_WSUM  (LDS_IDX + ROWS_PER_WG * 4)
#define LDS_TOTAL (LDS_WSUM + WAVES * 4)

__global__ __launch_bounds__(THREADS)
void vq_main(const float* __restrict__ z_e, const float* __restrict__ emb,
             float* __restrict__ out, float* __restrict__ ws)
{
    extern __shared__ unsigned char smem[];
    float*    enorm = (float*)(smem + LDS_ENORM);     // 512 f32: ||e_k||^2 (exact f32)
    _Float16* Af    = (_Float16*)(smem + LDS_A);      // 128 x 264 f16 (this block's rows)
    _Float16* Bf    = (_Float16*)(smem + LDS_B);      // 128 x 264 f16 (codebook chunk)
    int*      idxs  = (int*)(smem + LDS_IDX);         // 128 argmin results
    float*    wsum  = (float*)(smem + LDS_WSUM);      // 8 per-wave loss partials

    const int tid  = threadIdx.x;
    const int lane = tid & 31;
    const int wave = tid >> 5;
    const int hh   = lane >> 4;       // which 16-lane half
    const int ln16 = lane & 15;
    const int rowbase = blockIdx.x * ROWS_PER_WG;

    // ---- exact f32 codebook norms (codebook is L2-resident) ----
    for (int e = tid; e < KCODES; e += THREADS) {
        const float4* p = (const float4*)(emb + (size_t)e * DDIM);
        float s = 0.f;
        #pragma unroll
        for (int j = 0; j < DDIM / 4; ++j) {
            float4 v = p[j];
            s += v.x * v.x + v.y * v.y + v.z * v.z + v.w * v.w;
        }
        enorm[e] = s;
    }

    // ---- stage A: this block's 128 rows, f32 -> f16 into LDS ----
    {
        const float4* src = (const float4*)(z_e + (size_t)rowbase * DDIM);
        #pragma unroll
        for (int i = 0; i < (ROWS_PER_WG * DDIM / 4) / THREADS; ++i) { // 32 iters
            int idx4 = tid + i * THREADS;
            int r  = idx4 >> 6;     // 64 float4 per row
            int c4 = idx4 & 63;
            float4 v = src[idx4];   // rows contiguous in global
            _Float16* d = Af + r * PITCH + c4 * 4;
            d[0] = (_Float16)v.x; d[1] = (_Float16)v.y;
            d[2] = (_Float16)v.z; d[3] = (_Float16)v.w;
        }
    }

    // Per-lane running argmin state: 8 accumulator slots (C-matrix rows v, v+8 per half)
    float mv[8];
    int   mi[8];
    #pragma unroll
    for (int v = 0; v < 8; ++v) { mv[v] = 3.4e38f; mi[v] = 0; }

    const int mrow = (wave << 4) + ln16;   // local A row this lane feeds into the A-fragment

    for (int ch = 0; ch < NCHUNK; ++ch) {
        __syncthreads();
        // ---- stage B chunk: codebook rows [ch*128, ch*128+128), f32 -> f16 ----
        {
            const float4* src = (const float4*)(emb + (size_t)ch * CHUNK_COLS * DDIM);
            #pragma unroll
            for (int i = 0; i < (CHUNK_COLS * DDIM / 4) / THREADS; ++i) {
                int idx4 = tid + i * THREADS;
                int r  = idx4 >> 6;
                int c4 = idx4 & 63;
                float4 v = src[idx4];
                _Float16* d = Bf + r * PITCH + c4 * 4;
                d[0] = (_Float16)v.x; d[1] = (_Float16)v.y;
                d[2] = (_Float16)v.z; d[3] = (_Float16)v.w;
            }
        }
        __syncthreads();

        // ---- wave computes dot(16 rows, 128 codes) via v_wmma_f32_16x16x32_f16 ----
        #pragma unroll
        for (int ct = 0; ct < 8; ++ct) {
            v8f acc = {0.f, 0.f, 0.f, 0.f, 0.f, 0.f, 0.f, 0.f};
            #pragma unroll
            for (int ks = 0; ks < 8; ++ks) {
                const _Float16* ap = Af + mrow * PITCH + ks * 32;
                const _Float16* bp = Bf + (ct * 16 + ln16) * PITCH + ks * 32;
                v16h a, b;
                // A 16x32 frag, this code's contiguous k order: VGPR j -> K=2j+16h, M=lane%16
                // B 32x16 frag, same k order as A: VGPR j -> K=2j+16h, N=lane%16
                #pragma unroll
                for (int j = 0; j < 8; ++j) {
                    int ka = 2 * j + 16 * hh;
                    a[2 * j]     = ap[ka];
                    a[2 * j + 1] = ap[ka + 1];
                    int kb = 2 * j + 16 * hh;
                    b[2 * j]     = bp[kb];
                    b[2 * j + 1] = bp[kb + 1];
                }
                acc = __builtin_amdgcn_wmma_f32_16x16x32_f16(
                    false, a, false, b, (short)0, acc, false, false);
            }
            // C/D layout: N = lane%16, M = v + 8*(lane/16). Score = ||e||^2 - 2*dot
            // (per-row ||z||^2 is constant -> dropped from argmin).
            int n = ch * CHUNK_COLS + ct * 16 + ln16;
            float en = enorm[n];
            #pragma unroll
            for (int v = 0; v < 8; ++v) {
                float s = en - 2.0f * acc[v];
                if (s < mv[v]) { mv[v] = s; mi[v] = n; }  // strict <: keeps first (lowest) index
            }
        }
    }

    // ---- cross-lane argmin within each 16-lane half (cols are partitioned mod 16) ----
    #pragma unroll
    for (int v = 0; v < 8; ++v) {
        float val = mv[v]; int idx = mi[v];
        #pragma unroll
        for (int off = 1; off < 16; off <<= 1) {
            float ov = __shfl_xor(val, off, 32);
            int   oi = __shfl_xor(idx, off, 32);
            if (ov < val || (ov == val && oi < idx)) { val = ov; idx = oi; }
        }
        mv[v] = val; mi[v] = idx;
    }
    if (ln16 == 0) {  // lanes 0 and 16 hold rows v and v+8 of this wave's stripe
        #pragma unroll
        for (int v = 0; v < 8; ++v) {
            int rloc = (wave << 4) + v + 8 * hh;
            idxs[rloc] = mi[v];
            out[IDX_OFF + rowbase + rloc] = (float)mi[v];
        }
    }
    __syncthreads();

    // ---- gather z_q, write both output copies, accumulate commit loss ----
    float lsum = 0.f;
    const float4* zsrc = (const float4*)(z_e + (size_t)rowbase * DDIM);
    float4* zq1 = (float4*)(out + ZQST_OFF) + (size_t)rowbase * (DDIM / 4);
    #pragma unroll 4
    for (int i = 0; i < (ROWS_PER_WG * DDIM / 4) / THREADS; ++i) {
        int idx4 = tid + i * THREADS;
        int r  = idx4 >> 6;
        int c4 = idx4 & 63;
        int e  = idxs[r];
        float4 q = ((const float4*)(emb + (size_t)e * DDIM))[c4];
        float4 z = zsrc[idx4];
        float dx = z.x - q.x, dy = z.y - q.y, dz = z.z - q.z, dw = z.w - q.w;
        lsum += dx * dx + dy * dy + dz * dz + dw * dw;
        zq1[idx4] = q;  // z_q_st == z_q numerically (straight-through)
        // z_q_detached region starts at odd element offset -> only 4B aligned: scalar stores
        float* o2 = out + ZQD_OFF + (size_t)(rowbase + r) * DDIM + c4 * 4;
        o2[0] = q.x; o2[1] = q.y; o2[2] = q.z; o2[3] = q.w;
    }
    // deterministic per-wave then per-block loss partial
    #pragma unroll
    for (int off = 1; off < 32; off <<= 1) lsum += __shfl_xor(lsum, off, 32);
    if (lane == 0) wsum[wave] = lsum;
    __syncthreads();
    if (tid == 0) {
        float t = 0.f;
        #pragma unroll
        for (int w = 0; w < WAVES; ++w) t += wsum[w];
        ws[blockIdx.x] = t;
    }
}

__global__ __launch_bounds__(256)
void vq_loss_finalize(const float* __restrict__ ws, float* __restrict__ out)
{
    __shared__ float red[256];
    int tid = threadIdx.x;
    float s = 0.f;
    for (int i = tid; i < (N_ROWS / ROWS_PER_WG); i += 256) s += ws[i];  // fixed order
    red[tid] = s;
    __syncthreads();
    for (int k = 128; k > 0; k >>= 1) {
        if (tid < k) red[tid] += red[tid + k];
        __syncthreads();
    }
    if (tid == 0)
        out[LOSS_OFF] = red[0] * (0.25f / (float)((size_t)N_ROWS * DDIM));
}

extern "C" void kernel_launch(void* const* d_in, const int* in_sizes, int n_in,
                              void* d_out, int out_size, void* d_ws, size_t ws_size,
                              hipStream_t stream)
{
    const float* z_e = (const float*)d_in[0];   // (64, 2048, 256) f32
    const float* emb = (const float*)d_in[1];   // (512, 256) f32
    float* out = (float*)d_out;
    float* ws  = (float*)d_ws;

    vq_main<<<N_ROWS / ROWS_PER_WG, THREADS, LDS_TOTAL, stream>>>(z_e, emb, out, ws);
    vq_loss_finalize<<<1, 256, 0, stream>>>(ws, out);
}
